// CompressedModel_69295002353872
// MI455X (gfx1250) — matrix operand order repaired and checked by reading, not compiled, not run on hardware
//
#include <hip/hip_runtime.h>
#include <hip/hip_bf16.h>

typedef __attribute__((ext_vector_type(16))) _Float16 v16h;
typedef __attribute__((ext_vector_type(8)))  _Float16 v8h;
typedef __attribute__((ext_vector_type(8)))  float    v8f;

// exact parameter type of the async-LDS builtins (per hipcc diagnostic):
// pointer to gcc-style vector of 4 ints, in global (as1) / LDS (as3) address spaces
typedef int vs4i __attribute__((vector_size(4 * sizeof(int))));
typedef __attribute__((address_space(1))) vs4i* gptr16;
typedef __attribute__((address_space(3))) vs4i* lptr16;

#define B_    8
#define T_    2048
#define C_    768
#define RR    102            // r = floor(T - T*0.95)
#define NM    (2*RR)         // 204 merge candidates
#define NP    (T_ - NM)      // 1844 protected
#define NOUT  (NP + RR)      // 1946 output rows
#define NK    (C_ / 32)      // 24 k-steps

#if defined(__has_builtin)
#if __has_builtin(__builtin_amdgcn_global_load_async_to_lds_b128)
#define HAVE_ASYNC_LDS 1
#endif
#endif

// 16-byte global -> LDS copy; async (ASYNCcnt) when available, sync fallback otherwise
__device__ __forceinline__ void cp16_g2l(const _Float16* g, _Float16* l) {
#ifdef HAVE_ASYNC_LDS
  __builtin_amdgcn_global_load_async_to_lds_b128(
      (gptr16)(unsigned long long)g,   // global ptr: numeric value is identical in as1
      (lptr16)l,                       // generic -> as3 addrspacecast
      0, 0);
#else
  *(v8h*)l = *(const v8h*)g;
#endif
}
__device__ __forceinline__ void wait_async_lds() {
#ifdef HAVE_ASYNC_LDS
  asm volatile("s_wait_asynccnt 0x0" ::: "memory");
#endif
}

// ---------------- kernel 1: L2 normalize rows, f32 -> f16 ----------------
__global__ void k_normalize(const float* __restrict__ x, _Float16* __restrict__ xn) {
  const int row = blockIdx.x;                       // 0..B*T-1
  const float* xr = x + (size_t)row * C_;
  _Float16* outr = xn + (size_t)row * C_;
  float v[3]; float s = 0.f;
#pragma unroll
  for (int i = 0; i < 3; ++i) { v[i] = xr[threadIdx.x + i * 256]; s += v[i] * v[i]; }
  __shared__ float red[256];
  red[threadIdx.x] = s; __syncthreads();
  for (int off = 128; off > 0; off >>= 1) {
    if (threadIdx.x < off) red[threadIdx.x] += red[threadIdx.x + off];
    __syncthreads();
  }
  const float rn = rsqrtf(red[0]);
#pragma unroll
  for (int i = 0; i < 3; ++i) outr[threadIdx.x + i * 256] = (_Float16)(v[i] * rn);
}

// ---------------- zero / reciprocal helpers ----------------
__global__ void k_zero(float* __restrict__ p, int n) {
  int i = blockIdx.x * 256 + threadIdx.x;
  if (i < n) p[i] = 0.f;
}
__global__ void k_recip(float* __restrict__ p, int n) {
  int i = blockIdx.x * 256 + threadIdx.x;
  if (i < n) p[i] = 1.0f / p[i];
}

// ---------------- WMMA GEMM: sim tile + fused softmax epilogue ----------------
// PASS 0: Z[t]  += sum_s exp(-sim[t,s])          (row sums)
// PASS 1: iso[s] += sum_t exp(-sim[t,s])*invZ[t] (column sums; zacc holds invZ)
template <int PASS>
__global__ void __launch_bounds__(256) k_sim(const _Float16* __restrict__ xn,
                                             float* __restrict__ zacc,
                                             float* __restrict__ iso) {
  const int b  = blockIdx.z;
  const int tm = blockIdx.y;
  const int tn = blockIdx.x;
  const int tid   = threadIdx.x;
  const int lane  = tid & 31;
  const int wv    = tid >> 5;
  const int waveM = wv >> 2;          // 0..1  -> 64-row band
  const int waveN = wv & 3;           // 0..3  -> 32-col band
  const int lhalf = lane & 15;
  const int lhi   = lane >> 4;        // 0/1 : 16-lane half group

  const _Float16* base = xn + (size_t)b * T_ * C_;
  const _Float16* arow = base + (size_t)(tm * 128) * C_;
  const _Float16* brow = base + (size_t)(tn * 128) * C_;

  // double-buffered padded LDS tiles: 128 rows x 32 halves, stride 40 halves (80B)
  __shared__ __align__(16) _Float16 As[2][128 * 40];
  __shared__ __align__(16) _Float16 Bs[2][128 * 40];

  // per-thread tile-load assignment: rows r0 and r0+64, fixed 16B segment
  const int r0  = tid >> 2;
  const int seg = tid & 3;
  const _Float16* gA = arow + (size_t)r0 * C_ + seg * 8;
  const _Float16* gB = brow + (size_t)r0 * C_ + seg * 8;
  const int loff = r0 * 40 + seg * 8;

  v8f acc[4][2];
#pragma unroll
  for (int m = 0; m < 4; ++m)
#pragma unroll
    for (int n = 0; n < 2; ++n)
#pragma unroll
      for (int i = 0; i < 8; ++i) acc[m][n][i] = 0.0f;

  // prologue: stage k-step 0 into buffer 0
  cp16_g2l(gA,            &As[0][loff]);
  cp16_g2l(gA + 64 * C_,  &As[0][loff + 64 * 40]);
  cp16_g2l(gB,            &Bs[0][loff]);
  cp16_g2l(gB + 64 * C_,  &Bs[0][loff + 64 * 40]);

  for (int it = 0; it < NK; ++it) {
    const int cur = it & 1;
    wait_async_lds();                 // my loads into `cur` complete
    __syncthreads();                  // everyone's `cur` visible; everyone done reading cur^1
    if (it + 1 < NK) {                // stage next k-step into the other buffer (overlaps WMMA)
      const int nb = cur ^ 1;
      const int kq = (it + 1) * 32;
      cp16_g2l(gA + kq,           &As[nb][loff]);
      cp16_g2l(gA + kq + 64 * C_, &As[nb][loff + 64 * 40]);
      cp16_g2l(gB + kq,           &Bs[nb][loff]);
      cp16_g2l(gB + kq + 64 * C_, &Bs[nb][loff + 64 * 40]);
    }

    // fragments per ISA 16-bit layouts:
    // A 16x32: lane<16 holds K {0..7,16..23}; lane>=16 holds K {8..15,24..31}
    // B 32x16: lane<16 holds K {0..7,16..23}; lane>=16 holds K {8..15,24..31}
    v16h afrag[4], bfrag[2];
#pragma unroll
    for (int n = 0; n < 2; ++n) {
      const int r  = waveN * 32 + n * 16 + lhalf;
      const int kb = lhi * 8;
      v8h lo = *(const v8h*)(&Bs[cur][r * 40 + kb]);
      v8h hi = *(const v8h*)(&Bs[cur][r * 40 + kb + 16]);
      bfrag[n] = __builtin_shufflevector(lo, hi, 0,1,2,3,4,5,6,7,8,9,10,11,12,13,14,15);
    }
#pragma unroll
    for (int m = 0; m < 4; ++m) {
      const int r  = waveM * 64 + m * 16 + lhalf;
      const int kb = lhi * 8;
      v8h lo = *(const v8h*)(&As[cur][r * 40 + kb]);
      v8h hi = *(const v8h*)(&As[cur][r * 40 + kb + 16]);
      afrag[m] = __builtin_shufflevector(lo, hi, 0,1,2,3,4,5,6,7,8,9,10,11,12,13,14,15);
    }
#pragma unroll
    for (int m = 0; m < 4; ++m)
#pragma unroll
      for (int n = 0; n < 2; ++n)
        acc[m][n] = __builtin_amdgcn_wmma_f32_16x16x32_f16(
            false, afrag[m], false, bfrag[n], (short)0, acc[m][n], false, false);
    __syncthreads();                  // all waves done reading `cur` before it is refilled
  }

  // C/D layout: value i in lane L -> row (i + 8*lhi), col lhalf of the 16x16 frag
  if (PASS == 0) {
#pragma unroll
    for (int m = 0; m < 4; ++m) {
      float r8[8];
#pragma unroll
      for (int i = 0; i < 8; ++i)
        r8[i] = __expf(-acc[m][0][i]) + __expf(-acc[m][1][i]);
#pragma unroll
      for (int i = 0; i < 8; ++i) {   // reduce across 16 lanes of the half-group
        float v = r8[i];
        v += __shfl_xor(v, 8, 32);
        v += __shfl_xor(v, 4, 32);
        v += __shfl_xor(v, 2, 32);
        v += __shfl_xor(v, 1, 32);
        r8[i] = v;
      }
      if (lhalf == 0) {
        const int rowbase = tm * 128 + waveM * 64 + m * 16 + lhi * 8;
#pragma unroll
        for (int i = 0; i < 8; ++i)
          atomicAdd(&zacc[(size_t)b * T_ + rowbase + i], r8[i]);
      }
    }
  } else {
    float colsum[2] = {0.f, 0.f};
#pragma unroll
    for (int m = 0; m < 4; ++m) {
      const int rowbase = tm * 128 + waveM * 64 + m * 16 + lhi * 8;
      float zinv[8];
#pragma unroll
      for (int i = 0; i < 8; ++i) zinv[i] = zacc[(size_t)b * T_ + rowbase + i];
#pragma unroll
      for (int n = 0; n < 2; ++n)
#pragma unroll
        for (int i = 0; i < 8; ++i)
          colsum[n] += __expf(-acc[m][n][i]) * zinv[i];
    }
#pragma unroll
    for (int n = 0; n < 2; ++n) {
      const float v = colsum[n] + __shfl_xor(colsum[n], 16, 32);
      if (lhi == 0) {
        const int col = tn * 128 + waveN * 32 + n * 16 + lhalf;
        atomicAdd(&iso[(size_t)b * T_ + col], v);
      }
    }
  }
}

// ---------------- per-batch bitonic argsort (ascending) of iso ----------------
__global__ void k_argsort(const float* __restrict__ iso, int* __restrict__ idx) {
  const int b = blockIdx.x;
  __shared__ float sv[T_];
  __shared__ int   si[T_];
  for (int i = threadIdx.x; i < T_; i += blockDim.x) {
    sv[i] = iso[(size_t)b * T_ + i]; si[i] = i;
  }
  __syncthreads();
  for (int k = 2; k <= T_; k <<= 1) {
    for (int j = k >> 1; j > 0; j >>= 1) {
      for (int i = threadIdx.x; i < T_; i += blockDim.x) {
        const int ixj = i ^ j;
        if (ixj > i) {
          const bool up = ((i & k) == 0);
          const float a = sv[i], c = sv[ixj];
          if (up ? (a > c) : (a < c)) {
            sv[i] = c; sv[ixj] = a;
            const int t = si[i]; si[i] = si[ixj]; si[ixj] = t;
          }
        }
      }
      __syncthreads();
    }
  }
  for (int i = threadIdx.x; i < T_; i += blockDim.x)
    idx[(size_t)b * T_ + i] = si[i];
}

// ---------------- dst_idx[a] = argmax_b  xn[a_idx[a]] . xn[b_idx[b]] ----------------
__global__ void k_dst(const _Float16* __restrict__ xn, const int* __restrict__ idx,
                      int* __restrict__ dst) {
  const int b = blockIdx.y;
  const int a = blockIdx.x;                 // 0..RR-1
  const int* id = idx + (size_t)b * T_;
  const int ai = id[2 * a];
  __shared__ float sc[128];
  __shared__ int   sb[128];
  float best = -1e30f; int bestb = 0x7fffffff;
  const int t = threadIdx.x;
  if (t < RR) {
    const int bi = id[2 * t + 1];
    const v8h* pa = (const v8h*)(xn + ((size_t)b * T_ + ai) * C_);
    const v8h* pb = (const v8h*)(xn + ((size_t)b * T_ + bi) * C_);
    float s = 0.f;
    for (int c = 0; c < C_ / 8; ++c) {
      v8h va = pa[c], vb = pb[c];
#pragma unroll
      for (int e = 0; e < 8; ++e) s += (float)va[e] * (float)vb[e];
    }
    best = s; bestb = t;
  }
  sc[t] = best; sb[t] = bestb;
  __syncthreads();
  for (int off = 64; off > 0; off >>= 1) {   // first-index tie-break (match jnp.argmax)
    if (t < off) {
      const float o = sc[t + off]; const int ob = sb[t + off];
      if (o > sc[t] || (o == sc[t] && ob < sb[t])) { sc[t] = o; sb[t] = ob; }
    }
    __syncthreads();
  }
  if (t == 0) dst[(size_t)b * RR + a] = sb[0];
}

// ---------------- merge + renormalize ----------------
__global__ void k_merge(const float* __restrict__ x, const float* __restrict__ iso,
                        const int* __restrict__ idx, const int* __restrict__ dst,
                        float* __restrict__ out) {
  const int b = blockIdx.y;
  const int row = blockIdx.x;               // 0..NOUT-1
  const int* id = idx + (size_t)b * T_;
  const float* isob = iso + (size_t)b * T_;
  const float* xb = x + (size_t)b * T_ * C_;
  float* o = out + ((size_t)b * NOUT + row) * C_;
  if (row < NP) {                           // protected: (x*iso)/iso == x
    const float* xr = xb + (size_t)id[NM + row] * C_;
#pragma unroll
    for (int i = 0; i < 3; ++i) o[threadIdx.x + i * 256] = xr[threadIdx.x + i * 256];
  } else {
    const int j  = row - NP;
    const int bj = id[2 * j + 1];
    const float w = isob[bj];
    const float* xr = xb + (size_t)bj * C_;
    float accv[3]; float s = w;
#pragma unroll
    for (int i = 0; i < 3; ++i) accv[i] = xr[threadIdx.x + i * 256] * w;
    for (int a = 0; a < RR; ++a) {
      if (dst[(size_t)b * RR + a] == j) {
        const int ai = id[2 * a];
        const float wa = isob[ai];
        const float* xa = xb + (size_t)ai * C_;
#pragma unroll
        for (int i = 0; i < 3; ++i) accv[i] += xa[threadIdx.x + i * 256] * wa;
        s += wa;
      }
    }
    const float inv = 1.f / s;
#pragma unroll
    for (int i = 0; i < 3; ++i) o[threadIdx.x + i * 256] = accv[i] * inv;
  }
}

extern "C" void kernel_launch(void* const* d_in, const int* in_sizes, int n_in,
                              void* d_out, int out_size, void* d_ws, size_t ws_size,
                              hipStream_t stream) {
  const float* x = (const float*)d_in[0];
  float* out = (float*)d_out;

  // workspace layout (256B-aligned)
  char* ws = (char*)d_ws;
  size_t off = 0;
  _Float16* xn = (_Float16*)(ws + off); off += (size_t)B_ * T_ * C_ * sizeof(_Float16);
  off = (off + 255) & ~(size_t)255;
  float* Z   = (float*)(ws + off);      off += (size_t)B_ * T_ * sizeof(float);
  float* iso = (float*)(ws + off);      off += (size_t)B_ * T_ * sizeof(float);   // contiguous with Z
  off = (off + 255) & ~(size_t)255;
  int* idx = (int*)(ws + off);          off += (size_t)B_ * T_ * sizeof(int);
  int* dst = (int*)(ws + off);          off += (size_t)B_ * RR * sizeof(int);
  (void)ws_size; (void)in_sizes; (void)n_in; (void)out_size;

  k_normalize<<<B_ * T_, 256, 0, stream>>>(x, xn);
  k_zero<<<(2 * B_ * T_ + 255) / 256, 256, 0, stream>>>(Z, 2 * B_ * T_);  // Z + iso
  k_sim<0><<<dim3(T_ / 128, T_ / 128, B_), 256, 0, stream>>>(xn, Z, iso);
  k_recip<<<(B_ * T_ + 255) / 256, 256, 0, stream>>>(Z, B_ * T_);        // Z := 1/Z
  k_sim<1><<<dim3(T_ / 128, T_ / 128, B_), 256, 0, stream>>>(xn, Z, iso);
  k_argsort<<<B_, 1024, 0, stream>>>(iso, idx);
  k_dst<<<dim3(RR, B_), 128, 0, stream>>>(xn, idx, dst);
  k_merge<<<dim3(NOUT, B_), 256, 0, stream>>>(x, iso, idx, dst, out);
}
